// MultiHeadSelfAttention_37271726194910
// MI455X (gfx1250) — matrix operand order repaired and checked by reading, compiled, not run on hardware
//
#include <hip/hip_runtime.h>
#include <hip/hip_bf16.h>

typedef __bf16 bf16_t;
typedef __attribute__((ext_vector_type(16))) bf16_t v16bf;
typedef __attribute__((ext_vector_type(8)))  bf16_t v8bf;
typedef __attribute__((ext_vector_type(8)))  float  v8f;
typedef __attribute__((ext_vector_type(4)))  unsigned v4u;
typedef __attribute__((ext_vector_type(8)))  unsigned v8u;

#define S_LEN 2048
#define D_DIM 1024
#define H_NUM 16
#define DK    64
#define MTOT  4096           // B*S
// -2*ln(10000)/64
#define ROPE_C (-0.28782313662425572f)

#define USE_TDM 1

__device__ __forceinline__ bf16_t to_bf16(float f) {
  union { float f; unsigned u; } x; x.f = f;
  unsigned r = (x.u + 0x7FFFu + ((x.u >> 16) & 1u)) >> 16;
  union { unsigned short s; bf16_t b; } y; y.s = (unsigned short)r;
  return y.b;
}

__device__ __forceinline__ v8f zero8() {
  v8f z = {0.f,0.f,0.f,0.f,0.f,0.f,0.f,0.f};
  return z;
}

__device__ __forceinline__ v16bf concat8(v8bf lo, v8bf hi) {
  return __builtin_shufflevector(lo, hi, 0,1,2,3,4,5,6,7,8,9,10,11,12,13,14,15);
}

// ---------------------------------------------------------------- convert
__global__ void f32_to_bf16_kernel(const float* __restrict__ src,
                                   bf16_t* __restrict__ dst, int n) {
  int i = blockIdx.x * blockDim.x + threadIdx.x;
  if (i < n) dst[i] = to_bf16(src[i]);
}

// ---------------------------------------------------------------- TDM
// Issue a 2D TDM load: tile of 128 rows x 32 bf16 cols from a row-major
// [rows x 1024] bf16 tensor into LDS, with TDM padding of 4 dwords after
// every 16 dwords so the LDS row stride is 80 bytes (= LDT elems).
// D# layout per CDNA5 ISA ch.8 (group0: 4 SGPRs, group1: 8 SGPRs).
__device__ __forceinline__ void tdm_load_tile(unsigned ldsAddr,
                                              const bf16_t* gsrc,
                                              unsigned rows) {
  unsigned long long ga = (unsigned long long)(uintptr_t)gsrc;
  v4u g0;
  g0[0] = 1u;                                   // count=1, user mode
  g0[1] = ldsAddr;                              // lds_addr (bytes)
  g0[2] = (unsigned)(ga & 0xFFFFFFFFu);         // global_addr[31:0]
  g0[3] = (unsigned)((ga >> 32) & 0x01FFFFFFu)  // global_addr[56:32]
          | 0x80000000u;                        // type=2 ("image")
  v8u g1;
  g1[0] = (1u << 16)                            // data_size = 2 bytes
        | (1u << 20)                            // pad_enable
        | (3u << 22)                            // pad_interval: 16 dwords
        | (3u << 25);                           // pad_amount: 4 dwords
  g1[1] = (D_DIM & 0xFFFFu) << 16;              // tensor_dim0 lo16 @ [63:48]
  g1[2] = ((D_DIM >> 16) & 0xFFFFu)             // tensor_dim0 hi16 @ [79:64]
        | (rows & 0xFFFFu) << 16;               // tensor_dim1 lo16 @ [95:80]
  g1[3] = ((rows >> 16) & 0xFFFFu)              // tensor_dim1 hi16 @ [111:96]
        | (32u << 16);                          // tile_dim0 = 32  @ [127:112]
  g1[4] = 128u;                                 // tile_dim1 = 128, tile_dim2=0
  g1[5] = (unsigned)D_DIM;                      // tensor_dim0_stride lo32
  g1[6] = 0u;
  g1[7] = 0u;
  asm volatile("tensor_load_to_lds %0, %1" :: "s"(g0), "s"(g1) : "memory");
}

// ---------------------------------------------------------------- GEMM
// C[M,N] = A[M,K] * Bm[N,K]^T   (M=4096, N=K=1024), bf16 in, f32 accum.
// MODE 0: f32 row-major out (final projection)
// MODE 1: RoPE + bf16 out as [B,H,S,DK]       (Q, K)
// MODE 2: bf16 out transposed as [B,H,DK,S]   (V)
#define BM 128
#define BN 128
#define BK 32
#define LDT 40   // LDS row stride (bf16 elems); 80B rows keep 16B alignment

template <int MODE>
__global__ __launch_bounds__(256)
void gemm_bf16(const bf16_t* __restrict__ A, const bf16_t* __restrict__ Bm,
               const int* __restrict__ tokpos,
               float* __restrict__ Cf, bf16_t* __restrict__ Cb) {
  __shared__ bf16_t As[2][BM * LDT];
  __shared__ bf16_t Bs[2][BN * LDT];

  const int tid  = threadIdx.x;
  const int lane = tid & 31;
  const int wv   = tid >> 5;       // 0..7 waves
  const int wm0  = (wv & 3) * 32;  // 4 waves along M
  const int wn0  = (wv >> 2) * 64; // 2 waves along N
  const int m0   = blockIdx.y * BM;
  const int n0   = blockIdx.x * BN;
  const int half = lane >> 4;
  const int l16  = lane & 15;

  v8f c[2][4];
  #pragma unroll
  for (int i = 0; i < 2; ++i)
    #pragma unroll
    for (int j = 0; j < 4; ++j) c[i][j] = zero8();

#if USE_TDM
  const unsigned ldsA0 = (unsigned)(uintptr_t)&As[0][0];
  const unsigned ldsA1 = (unsigned)(uintptr_t)&As[1][0];
  const unsigned ldsB0 = (unsigned)(uintptr_t)&Bs[0][0];
  const unsigned ldsB1 = (unsigned)(uintptr_t)&Bs[1][0];

  // prologue: stage K-step 0 into buffer 0 (wave 0 drives the TDM)
  if (tid < 32) {
    tdm_load_tile(ldsA0, A  + (size_t)m0 * D_DIM, MTOT);
    tdm_load_tile(ldsB0, Bm + (size_t)n0 * D_DIM, D_DIM);
  }
  int cur = 0;
#else
  const int lrow = tid & 127;
  const int lcol = (tid >> 7) * 16;
#endif

  for (int k0 = 0; k0 < D_DIM; k0 += BK) {
#if USE_TDM
    // issue next tile's DMA into the other buffer, then wait for current.
    if (tid < 32) {
      if (k0 + BK < D_DIM) {
        tdm_load_tile(cur ? ldsA0 : ldsA1,
                      A + (size_t)m0 * D_DIM + k0 + BK, MTOT);
        tdm_load_tile(cur ? ldsB0 : ldsB1,
                      Bm + (size_t)n0 * D_DIM + k0 + BK, D_DIM);
        __builtin_amdgcn_s_wait_tensorcnt(2);  // current tile's 2 ops done
      } else {
        __builtin_amdgcn_s_wait_tensorcnt(0);
      }
    }
    __syncthreads();                           // data visible to all waves
    const bf16_t* __restrict__ Asb = &As[cur][0];
    const bf16_t* __restrict__ Bsb = &Bs[cur][0];
#else
    __syncthreads();
    {
      const v8bf* ga = reinterpret_cast<const v8bf*>(
          A + (size_t)(m0 + lrow) * D_DIM + k0 + lcol);
      v8bf a0 = ga[0], a1 = ga[1];
      *reinterpret_cast<v8bf*>(&As[0][lrow * LDT + lcol])     = a0;
      *reinterpret_cast<v8bf*>(&As[0][lrow * LDT + lcol + 8]) = a1;
      const v8bf* gb = reinterpret_cast<const v8bf*>(
          Bm + (size_t)(n0 + lrow) * D_DIM + k0 + lcol);
      v8bf b0 = gb[0], b1 = gb[1];
      *reinterpret_cast<v8bf*>(&Bs[0][lrow * LDT + lcol])     = b0;
      *reinterpret_cast<v8bf*>(&Bs[0][lrow * LDT + lcol + 8]) = b1;
    }
    __syncthreads();
    const bf16_t* __restrict__ Asb = &As[0][0];
    const bf16_t* __restrict__ Bsb = &Bs[0][0];
#endif

    // A fragments: 16x32, lane = row l16, K runs at half*8 and half*8+16
    v16bf af[2];
    #pragma unroll
    for (int mi = 0; mi < 2; ++mi) {
      const bf16_t* base = &Asb[(wm0 + mi * 16 + l16) * LDT + half * 8];
      v8bf lo = *reinterpret_cast<const v8bf*>(base);
      v8bf hi = *reinterpret_cast<const v8bf*>(base + 16);
      af[mi] = concat8(lo, hi);
    }
    // B fragments: 32x16, lane = column l16, K runs at half*8 and half*8+16
    v16bf bfr[4];
    #pragma unroll
    for (int ni = 0; ni < 4; ++ni) {
      const bf16_t* base = &Bsb[(wn0 + ni * 16 + l16) * LDT + half * 8];
      v8bf lo = *reinterpret_cast<const v8bf*>(base);
      v8bf hi = *reinterpret_cast<const v8bf*>(base + 16);
      bfr[ni] = concat8(lo, hi);
    }
    #pragma unroll
    for (int mi = 0; mi < 2; ++mi)
      #pragma unroll
      for (int ni = 0; ni < 4; ++ni)
        c[mi][ni] = __builtin_amdgcn_wmma_f32_16x16x32_bf16(
            false, af[mi], false, bfr[ni], (short)0, c[mi][ni], false, false);

#if USE_TDM
    __syncthreads();   // all reads of buf[cur] done before it is re-filled
    cur ^= 1;
#endif
  }

  // ------------------------------------------------------------- epilogue
  #pragma unroll
  for (int mi = 0; mi < 2; ++mi) {
    #pragma unroll
    for (int ni = 0; ni < 4; ++ni) {
      #pragma unroll
      for (int r = 0; r < 8; ++r) {
        float v = c[mi][ni][r];
        const int m = m0 + wm0 + mi * 16 + r + half * 8;
        const int n = n0 + wn0 + ni * 16 + l16;
        if (MODE == 0) {
          Cf[(size_t)m * D_DIM + n] = v;
        } else if (MODE == 1) {
          // RoPE: pair partner sits in adjacent lane (n differs by 1)
          float partner = __shfl_xor(v, 1, 32);
          const int sIdx = m & (S_LEN - 1);
          const int pos  = tokpos[sIdx];
          const int hd   = n & (DK - 1);
          const int j    = hd >> 1;
          float ang = (float)pos * __expf(ROPE_C * (float)j);
          float cs = __cosf(ang), sn = __sinf(ang);
          float y = (hd & 1) ? (partner * sn + v * cs)
                             : (v * cs - partner * sn);
          const int b = m >> 11;          // /S
          const int h = n >> 6;           // /DK
          Cb[(((size_t)(b * H_NUM + h) * S_LEN) + sIdx) * DK + hd] = to_bf16(y);
        } else {  // MODE 2: V transposed [B,H,DK,S]
          const int sIdx = m & (S_LEN - 1);
          const int b = m >> 11;
          const int h = n >> 6;
          const int hd = n & (DK - 1);
          Cb[(((size_t)(b * H_NUM + h) * DK) + hd) * S_LEN + sIdx] = to_bf16(v);
        }
      }
    }
  }
}

// ---------------------------------------------------------------- attention
// One wave handles a 16-query tile; flash-style online softmax over 32-key
// chunks. Q,K in [B,H,S,DK] bf16; V in [B,H,DK,S] bf16; out bf16 [B*S, D].
__global__ __launch_bounds__(256)
void attn_kernel(const bf16_t* __restrict__ Q, const bf16_t* __restrict__ K,
                 const bf16_t* __restrict__ Vt, bf16_t* __restrict__ O) {
  __shared__ bf16_t P[8][16 * 32];   // per-wave P-tile scratch (C->A relayout)

  const int tid  = threadIdx.x;
  const int lane = tid & 31;
  const int wv   = tid >> 5;
  const int half = lane >> 4;
  const int l16  = lane & 15;
  const int bh   = blockIdx.x >> 4;          // 0..31  (B*H)
  const int qblk = blockIdx.x & 15;
  const int qbase = qblk * 128 + wv * 16;
  const int b = bh >> 4, h = bh & 15;

  const bf16_t* Qh = Q  + (size_t)bh * S_LEN * DK;
  const bf16_t* Kh = K  + (size_t)bh * S_LEN * DK;
  const bf16_t* Vh = Vt + (size_t)bh * DK * S_LEN;

  // Q A-fragments, d=0..31 and d=32..63
  v16bf qa[2];
  #pragma unroll
  for (int t = 0; t < 2; ++t) {
    const bf16_t* base = Qh + (size_t)(qbase + l16) * DK + t * 32 + half * 8;
    v8bf lo = *reinterpret_cast<const v8bf*>(base);
    v8bf hi = *reinterpret_cast<const v8bf*>(base + 16);
    qa[t] = concat8(lo, hi);
  }

  v8f o[4];
  #pragma unroll
  for (int t = 0; t < 4; ++t) o[t] = zero8();
  float mrun[8], lrun[8];
  #pragma unroll
  for (int r = 0; r < 8; ++r) { mrun[r] = -1e30f; lrun[r] = 0.0f; }

  const int kend = qbase + 16;
  for (int kk = 0; kk < kend; kk += 32) {
    if (kk + 32 < kend) {   // prefetch next key chunk
      __builtin_prefetch(Kh + (size_t)(kk + 32 + l16) * DK, 0, 1);
      __builtin_prefetch(Vh + (size_t)l16 * S_LEN + kk + 32, 0, 1);
    }
    // ---- scores: two 16-key tiles, each = 2 WMMAs over d_k=64
    v8f sc[2];
    #pragma unroll
    for (int t = 0; t < 2; ++t) {
      v16bf bk[2];
      #pragma unroll
      for (int u = 0; u < 2; ++u) {
        const bf16_t* base =
            Kh + (size_t)(kk + t * 16 + l16) * DK + u * 32 + half * 8;
        v8bf lo = *reinterpret_cast<const v8bf*>(base);
        v8bf hi = *reinterpret_cast<const v8bf*>(base + 16);
        bk[u] = concat8(lo, hi);
      }
      v8f z = zero8();
      z = __builtin_amdgcn_wmma_f32_16x16x32_bf16(false, qa[0], false, bk[0],
                                                  (short)0, z, false, false);
      z = __builtin_amdgcn_wmma_f32_16x16x32_bf16(false, qa[1], false, bk[1],
                                                  (short)0, z, false, false);
      sc[t] = z;
    }
    // ---- causal mask + online softmax (per-row stats in 8 regs)
    float alpha[8];
    #pragma unroll
    for (int r = 0; r < 8; ++r) {
      const int row  = qbase + r + half * 8;
      const int key0 = kk + l16, key1 = key0 + 16;
      float v0 = (key0 <= row) ? sc[0][r] * 0.125f : -1e30f;
      float v1 = (key1 <= row) ? sc[1][r] * 0.125f : -1e30f;
      float tm = fmaxf(v0, v1);
      #pragma unroll
      for (int off = 8; off >= 1; off >>= 1)
        tm = fmaxf(tm, __shfl_xor(tm, off, 32));
      float mnew = fmaxf(mrun[r], tm);
      alpha[r] = __expf(mrun[r] - mnew);
      float p0 = __expf(v0 - mnew), p1 = __expf(v1 - mnew);
      float ps = p0 + p1;
      #pragma unroll
      for (int off = 8; off >= 1; off >>= 1) ps += __shfl_xor(ps, off, 32);
      lrun[r] = lrun[r] * alpha[r] + ps;
      mrun[r] = mnew;
      const int prow = r + half * 8;
      P[wv][prow * 32 + l16]      = to_bf16(p0);
      P[wv][prow * 32 + 16 + l16] = to_bf16(p1);
    }
    #pragma unroll
    for (int t = 0; t < 4; ++t)
      #pragma unroll
      for (int r = 0; r < 8; ++r) o[t][r] *= alpha[r];

    // ---- re-read P as an A fragment (wave-internal LDS is in-order)
    v16bf pa;
    {
      const bf16_t* base = &P[wv][l16 * 32 + half * 8];
      v8bf lo = *reinterpret_cast<const v8bf*>(base);
      v8bf hi = *reinterpret_cast<const v8bf*>(base + 16);
      pa = concat8(lo, hi);
    }
    // ---- P(16x32) @ V(32x64): 4 WMMAs; V is transposed so cols are rows
    #pragma unroll
    for (int t = 0; t < 4; ++t) {
      v16bf vb;
      const bf16_t* base = Vh + (size_t)(t * 16 + l16) * S_LEN + kk + half * 8;
      v8bf lo = *reinterpret_cast<const v8bf*>(base);
      v8bf hi = *reinterpret_cast<const v8bf*>(base + 16);
      vb = concat8(lo, hi);
      o[t] = __builtin_amdgcn_wmma_f32_16x16x32_bf16(false, pa, false, vb,
                                                     (short)0, o[t], false, false);
    }
  }

  // ---- normalize and store bf16 [B*S, H*DK] for final projection
  #pragma unroll
  for (int t = 0; t < 4; ++t) {
    #pragma unroll
    for (int r = 0; r < 8; ++r) {
      const int row = qbase + r + half * 8;
      float val = o[t][r] / lrun[r];
      const size_t M = (size_t)b * S_LEN + row;
      const int col = h * DK + t * 16 + l16;
      O[M * D_DIM + col] = to_bf16(val);
    }
  }
}

// ---------------------------------------------------------------- launch
extern "C" void kernel_launch(void* const* d_in, const int* in_sizes, int n_in,
                              void* d_out, int out_size, void* d_ws, size_t ws_size,
                              hipStream_t stream) {
  const float* x      = (const float*)d_in[0];
  const int*   tokpos = (const int*)d_in[1];
  const float* Wq     = (const float*)d_in[2];
  const float* Wk     = (const float*)d_in[3];
  const float* Wv     = (const float*)d_in[4];
  const float* Wo     = (const float*)d_in[5];
  float* out = (float*)d_out;

  const size_t NX = (size_t)MTOT * D_DIM;            // 4M
  const size_t NW = (size_t)D_DIM * D_DIM;           // 1M
  const size_t NQ = (size_t)2 * H_NUM * S_LEN * DK;  // 4M

  bf16_t* Xb  = (bf16_t*)d_ws;
  bf16_t* Wqb = Xb  + NX;
  bf16_t* Wkb = Wqb + NW;
  bf16_t* Wvb = Wkb + NW;
  bf16_t* Wob = Wvb + NW;
  bf16_t* Qb  = Wob + NW;
  bf16_t* Kb  = Qb  + NQ;
  bf16_t* Vtb = Kb  + NQ;
  bf16_t* Ob  = Vtb + NQ;

  f32_to_bf16_kernel<<<(int)((NX + 255) / 256), 256, 0, stream>>>(x,  Xb,  (int)NX);
  f32_to_bf16_kernel<<<(int)((NW + 255) / 256), 256, 0, stream>>>(Wq, Wqb, (int)NW);
  f32_to_bf16_kernel<<<(int)((NW + 255) / 256), 256, 0, stream>>>(Wk, Wkb, (int)NW);
  f32_to_bf16_kernel<<<(int)((NW + 255) / 256), 256, 0, stream>>>(Wv, Wvb, (int)NW);
  f32_to_bf16_kernel<<<(int)((NW + 255) / 256), 256, 0, stream>>>(Wo, Wob, (int)NW);

  dim3 gg(D_DIM / BN, MTOT / BM);  // (8, 32)
  gemm_bf16<1><<<gg, 256, 0, stream>>>(Xb, Wqb, tokpos, nullptr, Qb);
  gemm_bf16<1><<<gg, 256, 0, stream>>>(Xb, Wkb, tokpos, nullptr, Kb);
  gemm_bf16<2><<<gg, 256, 0, stream>>>(Xb, Wvb, tokpos, nullptr, Vtb);

  attn_kernel<<<2 * H_NUM * (S_LEN / 128), 256, 0, stream>>>(Qb, Kb, Vtb, Ob);

  gemm_bf16<0><<<gg, 256, 0, stream>>>(Ob, Wob, tokpos, out, nullptr);
}
